// MFMoE_17514876633916
// MI455X (gfx1250) — matrix operand order repaired and checked by reading, not compiled, not run on hardware
//
#include <hip/hip_runtime.h>
#include <hip/hip_bf16.h>
#include <math.h>

// ---------------------------------------------------------------------------
// MI455X / gfx1250: wave32, WMMA 16x16x32 bf16 (fp32 accum)
// ---------------------------------------------------------------------------

typedef __bf16 bf16_t;
typedef __attribute__((ext_vector_type(16))) bf16_t v16bf;
typedef __attribute__((ext_vector_type(8)))  float  v8f;

// problem constants (from setup_inputs)
#define BB    8
#define NTOK  5376
#define CIN   384
#define CH    768
#define MTOT  (BB * NTOK)      // 43008 rows for both GEMMs

// GEMM tiling
#define BM  128
#define BN  128
#define BK  64
#define BKP 72   // padded K-stride (bf16): 144 B rows -> every b128 16B-aligned

union Frag  { v16bf v; uint4 q[2]; };
union Pack8 { uint4 q; bf16_t h[8]; };

__device__ __forceinline__ float gelu_erf(float v) {
    return 0.5f * v * (1.0f + erff(v * 0.70710678118654752f));
}

// ---------------------------------------------------------------------------
// 0) fp32 (K,N) weights -> bf16 transposed (N,K).  Tiny (0.6 MB each); doing
//    the transpose here removes all LDS scatter stores from the GEMM hot loop.
// ---------------------------------------------------------------------------
__global__ void cvt_transpose_kernel(const float* __restrict__ src,
                                     bf16_t* __restrict__ dst, int K, int N) {
    int i = blockIdx.x * blockDim.x + threadIdx.x;
    if (i < N * K) {
        int n = i / K, k = i % K;
        dst[i] = (bf16_t)src[k * N + n];
    }
}

// ---------------------------------------------------------------------------
// 1/5) Tiled GEMM: C(MxN) = A(MxK) * B + bias (B given transposed, [N][K] bf16)
//      Block: 256 thr = 8 waves; wave grid 4(M) x 2(N); wave tile 32x64.
//      BK=64: 16 WMMAs per barrier pair.  Direct global->LDS loads (no
//      cross-barrier register staging: CU-mode forces 4 waves/SIMD for a
//      256-thread block, capping VGPRs ~136 -- staging spilled to scratch).
//      Latency is hidden by multi-block residency (36 KB LDS/block).
// ---------------------------------------------------------------------------
template <bool A_BF16>
__global__ void __launch_bounds__(256)
gemm_wmma(const void* __restrict__ Aptr, const bf16_t* __restrict__ Bt,
          const float* __restrict__ bias, float* __restrict__ Cout,
          int M, int Nn, int K)
{
    __shared__ __align__(16) bf16_t As[BM * BKP];   // [m][k]
    __shared__ __align__(16) bf16_t Bs[BN * BKP];   // [n][k]

    const int tid   = threadIdx.x;
    const int lane  = tid & 31;
    const int wid   = tid >> 5;
    const int lrow  = lane & 15;
    const int lhalf = lane >> 4;
    const int waveM = wid & 3;       // 4 wave-rows of 32
    const int waveN = wid >> 2;      // 2 wave-cols of 64
    const int m0 = blockIdx.x * BM;
    const int n0 = blockIdx.y * BN;

    v8f acc[2][4];
    #pragma unroll
    for (int i = 0; i < 2; ++i)
        #pragma unroll
        for (int j = 0; j < 4; ++j)
            #pragma unroll
            for (int e = 0; e < 8; ++e) acc[i][j][e] = 0.0f;

    // cooperative-load mapping: one 64-byte bf16 run per thread per tile
    const int lr = tid >> 1;          // tile row 0..127 (A: m, B: n)
    const int lc = (tid & 1) * 32;    // k base {0,32}

    const float*  Af = (const float*)Aptr;
    const bf16_t* Ab = (const bf16_t*)Aptr;

    for (int kt = 0; kt < K; kt += BK) {
        // ---- A tile (BM x BK) -> LDS (fp32 converted in flight) ----
        {
            uint4* da = (uint4*)&As[lr * BKP + lc];
            if (A_BF16) {
                const uint4* p = (const uint4*)(Ab + (size_t)(m0 + lr) * K + kt + lc);
                uint4 t0 = p[0], t1 = p[1], t2 = p[2], t3 = p[3];
                da[0] = t0; da[1] = t1; da[2] = t2; da[3] = t3;
            } else {
                const float4* p = (const float4*)(Af + (size_t)(m0 + lr) * K + kt + lc);
                #pragma unroll
                for (int h = 0; h < 2; ++h) {
                    const float4 f0 = p[4 * h + 0];
                    const float4 f1 = p[4 * h + 1];
                    const float4 f2 = p[4 * h + 2];
                    const float4 f3 = p[4 * h + 3];
                    Pack8 t0, t1;
                    t0.h[0] = (bf16_t)f0.x; t0.h[1] = (bf16_t)f0.y;
                    t0.h[2] = (bf16_t)f0.z; t0.h[3] = (bf16_t)f0.w;
                    t0.h[4] = (bf16_t)f1.x; t0.h[5] = (bf16_t)f1.y;
                    t0.h[6] = (bf16_t)f1.z; t0.h[7] = (bf16_t)f1.w;
                    t1.h[0] = (bf16_t)f2.x; t1.h[1] = (bf16_t)f2.y;
                    t1.h[2] = (bf16_t)f2.z; t1.h[3] = (bf16_t)f2.w;
                    t1.h[4] = (bf16_t)f3.x; t1.h[5] = (bf16_t)f3.y;
                    t1.h[6] = (bf16_t)f3.z; t1.h[7] = (bf16_t)f3.w;
                    da[2 * h]     = t0.q;
                    da[2 * h + 1] = t1.q;
                }
            }
        }
        // ---- B tile (BN x BK, already [n][k]) -> LDS ----
        {
            const uint4* q = (const uint4*)(Bt + (size_t)(n0 + lr) * K + kt + lc);
            uint4 t0 = q[0], t1 = q[1], t2 = q[2], t3 = q[3];
            uint4* db = (uint4*)&Bs[lr * BKP + lc];
            db[0] = t0; db[1] = t1; db[2] = t2; db[3] = t3;
        }
        __syncthreads();

        #pragma unroll
        for (int kk = 0; kk < 2; ++kk) {       // two 16x16x32 K-steps per stage
            Frag fa[2], fb[4];
            #pragma unroll
            for (int i = 0; i < 2; ++i) {
                // A 16x32: half 0 -> K {0..7,16..23}, half 1 -> K {8..15,24..31}
                const bf16_t* ap =
                    &As[(waveM * 32 + i * 16 + lrow) * BKP + kk * 32 + lhalf * 8];
                fa[i].q[0] = *(const uint4*)ap;
                fa[i].q[1] = *(const uint4*)(ap + 16);
            }
            #pragma unroll
            for (int j = 0; j < 4; ++j) {
                // B 32x16: lane = column, half 0 -> K {0..7,16..23}, half 1 -> K {8..15,24..31}
                const bf16_t* bp =
                    &Bs[(waveN * 64 + j * 16 + lrow) * BKP + kk * 32 + lhalf * 8];
                fb[j].q[0] = *(const uint4*)bp;
                fb[j].q[1] = *(const uint4*)(bp + 16);
            }
            #pragma unroll
            for (int i = 0; i < 2; ++i)
                #pragma unroll
                for (int j = 0; j < 4; ++j)
                    acc[i][j] = __builtin_amdgcn_wmma_f32_16x16x32_bf16(
                        false, fa[i].v, false, fb[j].v, (short)0, acc[i][j],
                        false, false);
        }
        __syncthreads();
    }

    // ---- epilogue: bias + fp32 store (D layout: VGPR v -> M = v + 8*half) ----
    #pragma unroll
    for (int j = 0; j < 4; ++j) {
        const int col = n0 + waveN * 64 + j * 16 + lrow;
        const float bv = bias[col];
        #pragma unroll
        for (int i = 0; i < 2; ++i) {
            #pragma unroll
            for (int v = 0; v < 8; ++v) {
                const int row = m0 + waveM * 32 + i * 16 + v + lhalf * 8;
                Cout[(size_t)row * Nn + col] = acc[i][j][v] + bv;
            }
        }
    }
}

// ---------------------------------------------------------------------------
// 2) depthwise 3x3 (SAME, cross-correlation) + bias, in place on h (B,N,C),
//    fused with avg/max pooling. One block per (scale, b, c) plane via LDS.
// ---------------------------------------------------------------------------
__global__ void __launch_bounds__(256)
dwconv_pool_kernel(float* __restrict__ hbuf,
                   const float* __restrict__ dw_w, const float* __restrict__ dw_b,
                   float* __restrict__ avgP, float* __restrict__ mxP)
{
    __shared__ float plane[4096];
    __shared__ float redS[256];
    __shared__ float redM[256];

    const int pid = blockIdx.x;
    const int s   = pid / (BB * CH);
    const int rem = pid % (BB * CH);
    const int b   = rem / CH;
    const int c   = rem % CH;

    const int dims[3] = {64, 32, 16};
    const int offs[3] = {0, 4096, 5120};
    const int hh = dims[s], ww = dims[s], base = offs[s];
    const int sp = hh * ww;                 // 4096 / 1024 / 256, all %256==0
    const int tid = threadIdx.x;

    const size_t gbase = ((size_t)b * NTOK + base) * CH + c;
    for (int idx = tid; idx < sp; idx += 256)
        plane[idx] = hbuf[gbase + (size_t)idx * CH];
    __syncthreads();

    float w[9];
    #pragma unroll
    for (int i = 0; i < 9; ++i) w[i] = dw_w[c * 9 + i];
    const float bval = dw_b[c];

    float ssum = 0.0f, smax = -3.4e38f;
    for (int idx = tid; idx < sp; idx += 256) {
        const int y = idx / ww, x = idx % ww;
        float acc = bval;
        #pragma unroll
        for (int dy = 0; dy < 3; ++dy) {
            const int yy = y + dy - 1;
            if (yy < 0 || yy >= hh) continue;
            #pragma unroll
            for (int dx = 0; dx < 3; ++dx) {
                const int xx = x + dx - 1;
                if (xx < 0 || xx >= ww) continue;
                acc += w[dy * 3 + dx] * plane[yy * ww + xx];
            }
        }
        hbuf[gbase + (size_t)idx * CH] = acc;   // in place (plane fully in LDS)
        ssum += acc;
        smax = fmaxf(smax, acc);
    }
    redS[tid] = ssum; redM[tid] = smax;
    __syncthreads();
    for (int off = 128; off > 0; off >>= 1) {
        if (tid < off) {
            redS[tid] += redS[tid + off];
            redM[tid]  = fmaxf(redM[tid], redM[tid + off]);
        }
        __syncthreads();
    }
    if (tid == 0) {
        const int pi = (s * BB + b) * CH + c;
        avgP[pi] = redS[0] / (float)sp;
        mxP[pi]  = redM[0];
    }
}

// ---------------------------------------------------------------------------
// 3) gating softmax + 3-tap channel convs -> per-(scale,b,c) multiplier
// ---------------------------------------------------------------------------
__global__ void __launch_bounds__(256)
gate_scale_kernel(const float* __restrict__ avgP, const float* __restrict__ mxP,
                  const float* __restrict__ gate_w, const float* __restrict__ gate_b,
                  const float* __restrict__ ca_w, const float* __restrict__ ra_w,
                  float* __restrict__ scaleS)
{
    __shared__ float pooled[CH];
    __shared__ float red0[256], red1[256];
    __shared__ float gwS[2];

    const int sb  = blockIdx.x;   // s*BB + b, 0..23
    const int tid = threadIdx.x;
    const float* avg = avgP + (size_t)sb * CH;
    const float* mx  = mxP  + (size_t)sb * CH;

    float d0 = 0.f, d1 = 0.f;
    for (int c = tid; c < CH; c += 256) {
        const float a = avg[c];
        pooled[c] = a + mx[c];
        d0 += a * gate_w[c * 2 + 0];
        d1 += a * gate_w[c * 2 + 1];
    }
    red0[tid] = d0; red1[tid] = d1;
    __syncthreads();
    for (int off = 128; off > 0; off >>= 1) {
        if (tid < off) { red0[tid] += red0[tid + off]; red1[tid] += red1[tid + off]; }
        __syncthreads();
    }
    if (tid == 0) {
        const float l0 = red0[0] + gate_b[0], l1 = red1[0] + gate_b[1];
        const float m  = fmaxf(l0, l1);
        const float e0 = __expf(l0 - m), e1 = __expf(l1 - m);
        const float inv = 1.0f / (e0 + e1);
        gwS[0] = e0 * inv; gwS[1] = e1 * inv;
    }
    __syncthreads();

    const float c0 = ca_w[0], c1 = ca_w[1], c2 = ca_w[2];
    const float r0 = ra_w[0], r1 = ra_w[1], r2 = ra_w[2];
    const float g0 = gwS[0], g1 = gwS[1];
    for (int c = tid; c < CH; c += 256) {
        const float pl = (c > 0)      ? pooled[c - 1] : 0.f;
        const float pc = pooled[c];
        const float pr = (c < CH - 1) ? pooled[c + 1] : 0.f;
        const float cav = c0 * pl + c1 * pc + c2 * pr;
        const float rav = r0 * pl + r1 * pc + r2 * pr;
        const float cas = 1.f / (1.f + __expf(-cav));
        const float ras = 1.f - 1.f / (1.f + __expf(-rav));
        scaleS[(size_t)sb * CH + c] = g0 * cas + g1 * ras;
    }
}

// ---------------------------------------------------------------------------
// 4) apply channel scale + exact-erf GELU, emit bf16 A-matrix for fc2
// ---------------------------------------------------------------------------
__global__ void __launch_bounds__(256)
scale_gelu_kernel(const float* __restrict__ hbuf, const float* __restrict__ scaleS,
                  bf16_t* __restrict__ g)
{
    const size_t i4   = (size_t)blockIdx.x * blockDim.x + threadIdx.x;
    const size_t flat = i4 * 4;
    const int    c  = (int)(flat % CH);
    const size_t bn = flat / CH;
    const int    n  = (int)(bn % NTOK);
    const int    b  = (int)(bn / NTOK);
    const int    s  = (n < 4096) ? 0 : ((n < 5120) ? 1 : 2);

    const float* sc = scaleS + ((size_t)(s * BB + b)) * CH + c;
    const float4 hv = *(const float4*)(hbuf + flat);

    union { uint2 u; bf16_t h[4]; } o;
    o.h[0] = (bf16_t)gelu_erf(hv.x * sc[0]);
    o.h[1] = (bf16_t)gelu_erf(hv.y * sc[1]);
    o.h[2] = (bf16_t)gelu_erf(hv.z * sc[2]);
    o.h[3] = (bf16_t)gelu_erf(hv.w * sc[3]);
    *(uint2*)(g + flat) = o.u;
}

// ---------------------------------------------------------------------------
// host side
// ---------------------------------------------------------------------------
extern "C" void kernel_launch(void* const* d_in, const int* in_sizes, int n_in,
                              void* d_out, int out_size, void* d_ws, size_t ws_size,
                              hipStream_t stream)
{
    const float* x      = (const float*)d_in[0];
    const float* fc1_w  = (const float*)d_in[1];
    const float* fc1_b  = (const float*)d_in[2];
    const float* dw_w   = (const float*)d_in[3];
    const float* dw_b   = (const float*)d_in[4];
    const float* ca_w   = (const float*)d_in[5];
    const float* ra_w   = (const float*)d_in[6];
    const float* gate_w = (const float*)d_in[7];
    const float* gate_b = (const float*)d_in[8];
    const float* fc2_w  = (const float*)d_in[9];
    const float* fc2_b  = (const float*)d_in[10];
    float* out = (float*)d_out;

    char*  ws  = (char*)d_ws;
    size_t off = 0;
    auto alloc = [&](size_t bytes) -> void* {
        off = (off + 255) & ~(size_t)255;
        void* p = ws + off;
        off += bytes;
        return p;
    };
    bf16_t* w1t  = (bf16_t*)alloc((size_t)CIN * CH * sizeof(bf16_t)); // [CH][CIN]
    bf16_t* w2t  = (bf16_t*)alloc((size_t)CH * CIN * sizeof(bf16_t)); // [CIN][CH]
    float*  hbuf = (float*) alloc((size_t)MTOT * CH * sizeof(float)); // 132 MB, L2-resident
    bf16_t* gbuf = (bf16_t*)alloc((size_t)MTOT * CH * sizeof(bf16_t));// 66 MB
    float*  avgP = (float*) alloc((size_t)3 * BB * CH * sizeof(float));
    float*  mxP  = (float*) alloc((size_t)3 * BB * CH * sizeof(float));
    float*  scS  = (float*) alloc((size_t)3 * BB * CH * sizeof(float));

    const int nw = CIN * CH;
    cvt_transpose_kernel<<<(nw + 255) / 256, 256, 0, stream>>>(fc1_w, w1t, CIN, CH);
    cvt_transpose_kernel<<<(nw + 255) / 256, 256, 0, stream>>>(fc2_w, w2t, CH, CIN);

    // fc1: (43008 x 384) * (384 x 768) + b
    gemm_wmma<false><<<dim3(MTOT / BM, CH / BN), 256, 0, stream>>>(
        x, w1t, fc1_b, hbuf, MTOT, CH, CIN);

    // depthwise conv + pooled stats (in place on hbuf)
    dwconv_pool_kernel<<<3 * BB * CH, 256, 0, stream>>>(hbuf, dw_w, dw_b, avgP, mxP);

    // gating + channel-attention multipliers
    gate_scale_kernel<<<3 * BB, 256, 0, stream>>>(avgP, mxP, gate_w, gate_b,
                                                  ca_w, ra_w, scS);

    // scale * GELU -> bf16
    scale_gelu_kernel<<<(int)(((size_t)MTOT * CH) / 1024), 256, 0, stream>>>(
        hbuf, scS, gbuf);

    // fc2: (43008 x 768) * (768 x 384) + b
    gemm_wmma<true><<<dim3(MTOT / BM, CIN / BN), 256, 0, stream>>>(
        gbuf, w2t, fc2_b, out, MTOT, CIN, CH);
}
